// LSTM_noforget_9560597201286
// MI455X (gfx1250) — matrix operand order repaired and checked by reading, not compiled, not run on hardware
//
#include <hip/hip_runtime.h>
#include <hip/hip_bf16.h>

// ---------------------------------------------------------------------------
// LSTM (no forget gate) on gfx1250 / MI455X, wave32 + WMMA f16 hi/lo (fp32-ish)
//   gates = [x_t, h_{t-1}] @ [W;U] + bias        (K = 256+512 = 768)
//   i = sigmoid, g = tanh, o = sigmoid ; c = i*g ; h = o*tanh(c)
// d_out layout (flat, fp32): out_seq(B,S,O) | hseq(B,S,H) | h_last(B,H) |
//                            c_last(B,H) | scalar 0
//
// Round 2: x is pre-split into f16 hi/lo planes by a parallel pre-pass so the
// sequential scan does no fp32->f16 conversion VALU on its critical path.
// ---------------------------------------------------------------------------

typedef __attribute__((ext_vector_type(16))) _Float16 v16h;
typedef __attribute__((ext_vector_type(8)))  float    v8f;

#define B_ 64
#define S_ 512
#define I_ 256
#define H_ 512
#define O_ 128
#define K_ 768            // I_ + H_
#define G3_ 1536          // 3*H_

#define OFF_HSEQ  ((size_t)B_ * S_ * O_)                  // 4194304
#define OFF_HLAST (OFF_HSEQ + (size_t)B_ * S_ * H_)       // 20971520
#define OFF_CLAST (OFF_HLAST + (size_t)B_ * H_)           // 21004288

static __device__ __forceinline__ v8f wmma_f16(v16h a, v16h b, v8f c) {
  // (neg_a, A, neg_b, B, c_mod, C, reuse_a, reuse_b)
  return __builtin_amdgcn_wmma_f32_16x16x32_f16(false, a, false, b, (short)0, c,
                                                false, false);
}

// A-matrix 16x32 f16 fragment (per ISA: lane m=lane&15, half=lane>>4;
// v16h elems 0..7 = K[half*8 .. +8), elems 8..15 = K[16+half*8 .. +8))
static __device__ __forceinline__ v16h frag_a_f16(const _Float16* row, int kb, int half) {
  const _Float16* p = row + kb + half * 8;
  v16h f;
#pragma unroll
  for (int i = 0; i < 8; ++i) { f[i] = p[i]; f[8 + i] = p[16 + i]; }
  return f;
}

// A fragment from fp32 memory, split into f16 hi/lo planes on the fly
// (used only in the fully-parallel readout GEMM, not on the scan path).
static __device__ __forceinline__ void frag_a_f32(const float* row, int kb, int half,
                                                  v16h& fh, v16h& fl) {
  const float* p = row + kb + half * 8;
#pragma unroll
  for (int i = 0; i < 8; ++i) {
    float a = p[i], b = p[16 + i];
    _Float16 ah = (_Float16)a, bh = (_Float16)b;
    fh[i]     = ah; fl[i]     = (_Float16)(a - (float)ah);
    fh[8 + i] = bh; fl[8 + i] = (_Float16)(b - (float)bh);
  }
}

// B-matrix 32x16 f16 fragment (lane n=lane&15, half=lane>>4; elems 0..7 =
// K[half*8 .. +8), elems 8..15 = K[16+half*8 .. +8)). Weights stored N-major.
static __device__ __forceinline__ v16h frag_b_f16(const _Float16* col, int kb, int half) {
  const _Float16* p = col + kb + half * 8;
  v16h f;
#pragma unroll
  for (int i = 0; i < 8; ++i) { f[i] = p[i]; f[8 + i] = p[16 + i]; }
  return f;
}

static __device__ __forceinline__ float sigm_f(float v) {
  return 1.f / (1.f + __expf(-v));
}
static __device__ __forceinline__ float tanh_f(float v) {
  float t = __expf(-2.f * fabsf(v));
  float r = (1.f - t) / (1.f + t);
  return copysignf(r, v);
}

// ---------------------------------------------------------------------------
// Kernel 0a: pack weights.  Vt[n][k] = (k<I_ ? W[k][n] : U[k-I_][n]) as f16
// hi/lo planes (n-major, K_=768 per row).  woT[o][h] = wo[h][o], same split.
// Also writes the trailing scalar 0 of d_out.
// ---------------------------------------------------------------------------
__global__ __launch_bounds__(256) void lstm_pack_weights(
    const float* __restrict__ W, const float* __restrict__ U,
    const float* __restrict__ wo,
    _Float16* __restrict__ VtH, _Float16* __restrict__ VtL,
    _Float16* __restrict__ WoH, _Float16* __restrict__ WoL,
    float* __restrict__ out, int out_size) {
  const int idx = blockIdx.x * 256 + threadIdx.x;
  const int NV = G3_ * K_;             // 1,179,648
  if (idx < NV) {
    const int n = idx / K_;
    const int k = idx - n * K_;
    float v = (k < I_) ? W[(size_t)k * G3_ + n] : U[(size_t)(k - I_) * G3_ + n];
    _Float16 h = (_Float16)v;
    VtH[idx] = h;
    VtL[idx] = (_Float16)(v - (float)h);
  } else if (idx < NV + O_ * H_) {
    const int r = idx - NV;
    const int o = r >> 9;              // / H_
    const int h = r & (H_ - 1);
    float v = wo[(size_t)h * O_ + o];
    _Float16 hh = (_Float16)v;
    WoH[r] = hh;
    WoL[r] = (_Float16)(v - (float)hh);
  }
  if (idx == 0) out[out_size - 1] = 0.f;   // the literal 0 in the output tuple
}

// ---------------------------------------------------------------------------
// Kernel 0b: pack x (B,S,I) fp32 -> f16 hi/lo planes (same linear layout).
// Fully parallel, bandwidth-trivial (~67 MB total at 23.3 TB/s).
// ---------------------------------------------------------------------------
__global__ __launch_bounds__(256) void lstm_pack_x(
    const float* __restrict__ x,
    _Float16* __restrict__ xH, _Float16* __restrict__ xL) {
  const size_t base = ((size_t)blockIdx.x * 256 + threadIdx.x) * 8;
#pragma unroll
  for (int i = 0; i < 8; ++i) {
    float v = x[base + i];
    _Float16 h = (_Float16)v;
    xH[base + i] = h;
    xL[base + i] = (_Float16)(v - (float)h);
  }
}

// ---------------------------------------------------------------------------
// Kernel 1: recurrent scan.  4 blocks x 256 threads; block owns batch rows
// [b0, b0+16).  h double-buffered in LDS as f16 hi/lo; one barrier per step.
// Each wave owns 4 h-column tiles (8 waves * 4 * 16 = 512 = H_).
// Hot loop is pure f16 fragment loads (global/LDS b128) + WMMA; the only
// VALU is the 8-element activation tail per tile.
// ---------------------------------------------------------------------------
__global__ __launch_bounds__(256) void lstm_scan(
    const _Float16* __restrict__ xH, const _Float16* __restrict__ xL,
    const float* __restrict__ bias,
    const _Float16* __restrict__ VtH, const _Float16* __restrict__ VtL,
    float* __restrict__ out) {
  __shared__ _Float16 hsH[2][16][H_];   // 32 KB
  __shared__ _Float16 hsL[2][16][H_];   // 32 KB

  const int b0   = blockIdx.x * 16;
  const int tid  = threadIdx.x;
  const int wave = tid >> 5;
  const int lane = tid & 31;
  const int half = lane >> 4;
  const int lm   = lane & 15;

  float* hseq  = out + OFF_HSEQ;
  float* hlast = out + OFF_HLAST;
  float* clast = out + OFF_CLAST;

  // h_0 = 0 in buffer 0 (covered by the barrier at top of the s-loop)
  for (int i = tid; i < 16 * H_; i += 256) {
    (&hsH[0][0][0])[i] = (_Float16)0.f;
    (&hsL[0][0][0])[i] = (_Float16)0.f;
  }

  // Per-lane A row = batch row b0+lm of x (f16 hi/lo planes).
  const _Float16* xrowH = xH + (size_t)(b0 + lm) * S_ * I_;
  const _Float16* xrowL = xL + (size_t)(b0 + lm) * S_ * I_;

  // Per-lane bias for the 4 owned column tiles (C/D element depends on n=lm only)
  float biA[4], bgA[4], boA[4];
#pragma unroll
  for (int t = 0; t < 4; ++t) {
    const int hc = (wave * 4 + t) * 16 + lm;
    biA[t] = bias[hc];
    bgA[t] = bias[H_ + hc];
    boA[t] = bias[2 * H_ + hc];
  }

#pragma unroll 1
  for (int s = 0; s < S_; ++s) {
    __syncthreads();                       // h_{s-1} (and init) visible
    const int rp = s & 1, wp = rp ^ 1;
    const _Float16* xrH = xrowH + (size_t)s * I_;
    const _Float16* xrL = xrowL + (size_t)s * I_;
    if (s + 1 < S_) {                      // prefetch next-step x rows
      __builtin_prefetch(xrH + I_, 0, 1);
      __builtin_prefetch(xrL + I_, 0, 1);
    }

    const _Float16* hrowH = &hsH[rp][lm][0];
    const _Float16* hrowL = &hsL[rp][lm][0];

#pragma unroll 1
    for (int t = 0; t < 4; ++t) {
      const int hc = (wave * 4 + t) * 16;          // absolute h column base
      const float bi = biA[t], bg = bgA[t], bo = boA[t];
      v8f accI = {bi, bi, bi, bi, bi, bi, bi, bi};
      v8f accG = {bg, bg, bg, bg, bg, bg, bg, bg};
      v8f accO = {bo, bo, bo, bo, bo, bo, bo, bo};

      const _Float16* BIH = VtH + (size_t)(hc + lm) * K_;
      const _Float16* BIL = VtL + (size_t)(hc + lm) * K_;
      const _Float16* BGH = BIH + (size_t)H_ * K_;
      const _Float16* BGL = BIL + (size_t)H_ * K_;
      const _Float16* BOH = BGH + (size_t)H_ * K_;
      const _Float16* BOL = BGL + (size_t)H_ * K_;

      // ---- x @ W part: K = 0..255 (A from global f16 hi/lo planes)
#pragma unroll 2
      for (int kc = 0; kc < 8; ++kc) {
        const int kb = kc * 32;
        v16h ah = frag_a_f16(xrH, kb, half);
        v16h al = frag_a_f16(xrL, kb, half);
        v16h bh = frag_b_f16(BIH, kb, half), bl = frag_b_f16(BIL, kb, half);
        accI = wmma_f16(ah, bh, accI);
        accI = wmma_f16(ah, bl, accI);
        accI = wmma_f16(al, bh, accI);
        bh = frag_b_f16(BGH, kb, half); bl = frag_b_f16(BGL, kb, half);
        accG = wmma_f16(ah, bh, accG);
        accG = wmma_f16(ah, bl, accG);
        accG = wmma_f16(al, bh, accG);
        bh = frag_b_f16(BOH, kb, half); bl = frag_b_f16(BOL, kb, half);
        accO = wmma_f16(ah, bh, accO);
        accO = wmma_f16(ah, bl, accO);
        accO = wmma_f16(al, bh, accO);
      }

      // ---- h @ U part: K = 256..767 (A from LDS f16 hi/lo planes)
#pragma unroll 2
      for (int kc = 0; kc < 16; ++kc) {
        const int kb = kc * 32;          // within h (0..480)
        const int kg = I_ + kb;          // within packed Vt rows
        v16h ah = frag_a_f16(hrowH, kb, half);
        v16h al = frag_a_f16(hrowL, kb, half);
        v16h bh = frag_b_f16(BIH, kg, half), bl = frag_b_f16(BIL, kg, half);
        accI = wmma_f16(ah, bh, accI);
        accI = wmma_f16(ah, bl, accI);
        accI = wmma_f16(al, bh, accI);
        bh = frag_b_f16(BGH, kg, half); bl = frag_b_f16(BGL, kg, half);
        accG = wmma_f16(ah, bh, accG);
        accG = wmma_f16(ah, bl, accG);
        accG = wmma_f16(al, bh, accG);
        bh = frag_b_f16(BOH, kg, half); bl = frag_b_f16(BOL, kg, half);
        accO = wmma_f16(ah, bh, accO);
        accO = wmma_f16(ah, bl, accO);
        accO = wmma_f16(al, bh, accO);
      }

      // ---- activations + h/c writes.  C/D element j -> (m=j+8*half, n=lm)
#pragma unroll
      for (int j = 0; j < 8; ++j) {
        const int m = j + half * 8;
        const int n = hc + lm;
        float iv = sigm_f(accI[j]);
        float gv = tanh_f(accG[j]);
        float ov = sigm_f(accO[j]);
        float c  = iv * gv;
        float h  = ov * tanh_f(c);
        _Float16 hh = (_Float16)h;
        hsH[wp][m][n] = hh;
        hsL[wp][m][n] = (_Float16)(h - (float)hh);
        hseq[((size_t)(b0 + m) * S_ + s) * H_ + n] = h;
        if (s == S_ - 1) {
          hlast[(size_t)(b0 + m) * H_ + n] = h;
          clast[(size_t)(b0 + m) * H_ + n] = c;
        }
      }
    }
  }
}

// ---------------------------------------------------------------------------
// Kernel 2: readout GEMM  out_seq = hseq @ wo.  M=B*S=32768, N=O_=128, K=H_.
// One wave per 16x16 tile; 8 waves/block -> 2048 blocks.  Converts hseq fp32
// to f16 hi/lo on the fly (fully parallel, off the critical path).
// ---------------------------------------------------------------------------
__global__ __launch_bounds__(256) void lstm_readout(
    const float* __restrict__ hseq, const _Float16* __restrict__ WoH,
    const _Float16* __restrict__ WoL, float* __restrict__ outseq) {
  const int tid  = threadIdx.x;
  const int wave = tid >> 5;
  const int lane = tid & 31;
  const int half = lane >> 4;
  const int lm   = lane & 15;

  const int tile = blockIdx.x * 8 + wave;     // 0..16383
  const int mt = tile >> 3;                   // 2048 M tiles
  const int nt = tile & 7;                    // 8 N tiles

  const float*    arow = hseq + (size_t)(mt * 16 + lm) * H_;
  const _Float16* bH   = WoH + (size_t)(nt * 16 + lm) * H_;
  const _Float16* bL   = WoL + (size_t)(nt * 16 + lm) * H_;

  v8f acc = {};
#pragma unroll 4
  for (int kc = 0; kc < 16; ++kc) {
    const int kb = kc * 32;
    v16h ah, al;
    frag_a_f32(arow, kb, half, ah, al);
    v16h bh = frag_b_f16(bH, kb, half);
    v16h bl = frag_b_f16(bL, kb, half);
    acc = wmma_f16(ah, bh, acc);
    acc = wmma_f16(ah, bl, acc);
    acc = wmma_f16(al, bh, acc);
  }
#pragma unroll
  for (int j = 0; j < 8; ++j) {
    outseq[(size_t)(mt * 16 + j + 8 * half) * O_ + nt * 16 + lm] = acc[j];
  }
}

// ---------------------------------------------------------------------------
extern "C" void kernel_launch(void* const* d_in, const int* in_sizes, int n_in,
                              void* d_out, int out_size, void* d_ws, size_t ws_size,
                              hipStream_t stream) {
  const float* x    = (const float*)d_in[0];   // (B,S,I)
  const float* W    = (const float*)d_in[1];   // (I,3H)
  const float* U    = (const float*)d_in[2];   // (H,3H)
  const float* bias = (const float*)d_in[3];   // (3H,)
  const float* wo   = (const float*)d_in[4];   // (H,O)
  float* out = (float*)d_out;

  // Workspace: f16 hi/lo planes of [W;U]^T (1536x768), wo^T (128x512),
  // and x (B,S,I).  Total ~38.5 MB.
  _Float16* VtH = (_Float16*)d_ws;
  _Float16* VtL = VtH + (size_t)G3_ * K_;
  _Float16* WoH = VtL + (size_t)G3_ * K_;
  _Float16* WoL = WoH + (size_t)O_ * H_;
  _Float16* xHp = WoL + (size_t)O_ * H_;
  _Float16* xLp = xHp + (size_t)B_ * S_ * I_;

  const int packN = G3_ * K_ + O_ * H_;        // 1,245,184 -> 4864 blocks
  lstm_pack_weights<<<(packN + 255) / 256, 256, 0, stream>>>(
      W, U, wo, VtH, VtL, WoH, WoL, out, out_size);

  const int xThreads = (B_ * S_ * I_) / 8;     // 1,048,576 threads
  lstm_pack_x<<<xThreads / 256, 256, 0, stream>>>(x, xHp, xLp);

  lstm_scan<<<4, 256, 0, stream>>>(xHp, xLp, bias, VtH, VtL, out);

  lstm_readout<<<2048, 256, 0, stream>>>(out + OFF_HSEQ, WoH, WoL, out);
}
